// Block_78993038508729
// MI455X (gfx1250) — matrix operand order repaired and checked by reading, not compiled, not run on hardware
//
#include <hip/hip_runtime.h>
#include <hip/hip_bf16.h>
#include <math.h>

#define TT 2048
#define DD 1024
#define NHEAD 16
#define HDIM 64
#define NE 8
#define HHID 2048

typedef __attribute__((ext_vector_type(16))) __bf16 v16bf;
typedef __attribute__((ext_vector_type(8)))  float  v8f;

union Frag {
  v16bf v;
  uint4 q[2];
  unsigned short u[16];
};

__device__ __forceinline__ unsigned short f2bf(float f) {
  unsigned int u = __float_as_uint(f);
  unsigned int r = u + 0x7FFFu + ((u >> 16) & 1u);
  return (unsigned short)(r >> 16);
}
__device__ __forceinline__ float bf2f(unsigned short h) {
  return __uint_as_float(((unsigned int)h) << 16);
}
__device__ __forceinline__ v8f wmma_bf16(const Frag& a, const Frag& b, v8f c) {
  return __builtin_amdgcn_wmma_f32_16x16x32_bf16(false, a.v, false, b.v,
                                                 (short)0, c, false, false);
}

// ---------------------------------------------------------------------------
// fp32 -> bf16 elementwise convert (weights, grid-stride)
// ---------------------------------------------------------------------------
__global__ __launch_bounds__(256) void f32_to_bf16_kernel(
    const float* __restrict__ in, unsigned short* __restrict__ out, size_t n) {
  size_t i = (size_t)blockIdx.x * 256 + threadIdx.x;
  size_t stride = (size_t)gridDim.x * 256;
  for (; i < n; i += stride) out[i] = f2bf(in[i]);
}

// ---------------------------------------------------------------------------
// RMSNorm: fp32 [rows, DD] -> bf16 [rows, DD]
// ---------------------------------------------------------------------------
__global__ __launch_bounds__(256) void rmsnorm_kernel(
    const float* __restrict__ x, const float* __restrict__ w,
    unsigned short* __restrict__ out) {
  int row = blockIdx.x;
  int tid = threadIdx.x;
  const float* xr = x + (size_t)row * DD;
  float s = 0.f;
  for (int i = tid; i < DD; i += 256) { float v = xr[i]; s += v * v; }
#pragma unroll
  for (int m = 16; m >= 1; m >>= 1) s += __shfl_xor(s, m, 32);
  __shared__ float wsum[8];
  if ((tid & 31) == 0) wsum[tid >> 5] = s;
  __syncthreads();
  float tot = 0.f;
#pragma unroll
  for (int j = 0; j < 8; ++j) tot += wsum[j];
  float inv = rsqrtf(tot / (float)DD + 1e-8f);
  for (int i = tid; i < DD; i += 256)
    out[(size_t)row * DD + i] = f2bf(xr[i] * inv * w[i]);
}

// ---------------------------------------------------------------------------
// Generic bf16 WMMA GEMM: C[M,N] (+)= epilogue( A[M,K](bf16) @ B[K,N](bf16) )
// 64x128 tile per 256-thread block; 8 waves as 2(M) x 4(N); each wave owns a
// 32x32 patch = 2 A-frags x 2 B-frags -> 4 WMMAs per 32-K step (2 LDS b128
// loads per WMMA, each fragment reused twice).
// SCALE_MODE: 0=none, 1=mask-only (skip tile, write zeros), 2=multiply+skip
// ---------------------------------------------------------------------------
template<int SCALE_MODE, bool ACCUM, bool BIAS, bool RES>
__global__ __launch_bounds__(256) void gemm_bf16(
    const unsigned short* __restrict__ A, const unsigned short* __restrict__ B,
    const float* __restrict__ bias, const float* __restrict__ Res,
    const float* __restrict__ rowscale, int rs_stride, float rs_mul,
    float* __restrict__ C, int M, int N, int K) {
  const int tid  = threadIdx.x;
  const int lane = tid & 31;
  const int wave = tid >> 5;
  const int row0 = blockIdx.y * 64;
  const int col0 = blockIdx.x * 128;

  __shared__ int anyNZ;
  if (SCALE_MODE != 0) {
    if (tid == 0) anyNZ = 0;
    __syncthreads();
    if (tid < 64) {
      if (rowscale[(size_t)(row0 + tid) * rs_stride] != 0.0f) anyNZ = 1;
    }
    __syncthreads();
    if (!anyNZ) {
      if (!ACCUM) {
        for (int i = tid; i < 64 * 128; i += 256)
          C[(size_t)(row0 + (i >> 7)) * N + col0 + (i & 127)] = 0.0f;
      }
      return;
    }
  }

  __shared__ __align__(16) unsigned short As[64 * 40];    // [m][k] tile
  __shared__ __align__(16) unsigned short Bs[128 * 40];   // B^T tile: [n][k]

  const int wm = wave >> 2;        // 0..1 : 32-row slab
  const int wn = wave & 3;         // 0..3 : 32-col slab
  const int hi = lane >> 4;        // 0..1
  const int l15 = lane & 15;
  const int koffA = 16 * hi;       // A operand K pattern
  const int koffB = 16 * hi;       // B operand K pattern

  v8f acc00 = {}, acc01 = {}, acc10 = {}, acc11 = {};

  const int arow = tid >> 2;       // 0..63
  const int ac8  = (tid & 3) * 8;  // 0,8,16,24
  const int brow = tid >> 3;       // 0..31
  const int bc16 = (tid & 7) * 16; // 0..112

  for (int k0 = 0; k0 < K; k0 += 32) {
    // A tile 64x32 (one b128 per thread)
    *(uint4*)&As[arow * 40 + ac8] =
        *(const uint4*)(A + (size_t)(row0 + arow) * K + k0 + ac8);
    // B tile 32x128 -> transposed into Bs[n][k] (two b128 reads per thread)
    uint4 bq0 = *(const uint4*)(B + (size_t)(k0 + brow) * N + col0 + bc16);
    uint4 bq1 = *(const uint4*)(B + (size_t)(k0 + brow) * N + col0 + bc16 + 8);
    const unsigned short* bh0 = (const unsigned short*)&bq0;
    const unsigned short* bh1 = (const unsigned short*)&bq1;
#pragma unroll
    for (int j = 0; j < 8; ++j) {
      Bs[(bc16 + j) * 40 + brow]     = bh0[j];
      Bs[(bc16 + 8 + j) * 40 + brow] = bh1[j];
    }
    __syncthreads();

    Frag b0, b1, a0, a1;
    b0.q[0] = *(const uint4*)&Bs[(wn * 32 + l15) * 40 + koffB];
    b0.q[1] = *(const uint4*)&Bs[(wn * 32 + l15) * 40 + koffB + 8];
    b1.q[0] = *(const uint4*)&Bs[(wn * 32 + 16 + l15) * 40 + koffB];
    b1.q[1] = *(const uint4*)&Bs[(wn * 32 + 16 + l15) * 40 + koffB + 8];
    a0.q[0] = *(const uint4*)&As[(wm * 32 + l15) * 40 + koffA];
    a0.q[1] = *(const uint4*)&As[(wm * 32 + l15) * 40 + koffA + 8];
    a1.q[0] = *(const uint4*)&As[(wm * 32 + 16 + l15) * 40 + koffA];
    a1.q[1] = *(const uint4*)&As[(wm * 32 + 16 + l15) * 40 + koffA + 8];
    acc00 = wmma_bf16(a0, b0, acc00);
    acc01 = wmma_bf16(a0, b1, acc01);
    acc10 = wmma_bf16(a1, b0, acc10);
    acc11 = wmma_bf16(a1, b1, acc11);
    __syncthreads();
  }

#pragma unroll
  for (int mf = 0; mf < 2; ++mf) {
#pragma unroll
    for (int nf = 0; nf < 2; ++nf) {
      v8f a = mf ? (nf ? acc11 : acc10) : (nf ? acc01 : acc00);
      const int gn = col0 + wn * 32 + nf * 16 + l15;
#pragma unroll
      for (int r = 0; r < 8; ++r) {
        int gm = row0 + wm * 32 + mf * 16 + 8 * hi + r;
        float v = a[r];
        if (BIAS) v += bias[gn];
        if (SCALE_MODE == 2) v *= rowscale[(size_t)gm * rs_stride] * rs_mul;
        if (RES) v += Res[(size_t)gm * N + gn];
        size_t idx = (size_t)gm * N + gn;
        if (ACCUM) v += C[idx];
        C[idx] = v;
      }
    }
  }
}

// ---------------------------------------------------------------------------
// RoPE-as-source + pack q/k/v into per-head bf16 [NHEAD][TT][HDIM].
// Rotation angle indexed by HEAD (faithful to the reference's quirk).
// ---------------------------------------------------------------------------
__global__ __launch_bounds__(256) void rope_pack_kernel(
    const float* __restrict__ qf, const float* __restrict__ kvf,
    const float* __restrict__ fcos, const float* __restrict__ fsin,
    unsigned short* __restrict__ Qb, unsigned short* __restrict__ Kb,
    unsigned short* __restrict__ Vb) {
  size_t gid = (size_t)blockIdx.x * 256 + threadIdx.x;  // TT*NHEAD*(HDIM/2)
  int i  = gid & 31;
  int hh = (gid >> 5) & 15;
  int t  = (int)(gid >> 9);
  float c = fcos[hh * (HDIM / 2) + i];
  float s = fsin[hh * (HDIM / 2) + i];
  size_t dst = ((size_t)hh * TT + t) * HDIM + 2 * i;

  size_t qsrc = (size_t)t * DD + hh * HDIM + 2 * i;
  float a = qf[qsrc], b = qf[qsrc + 1];
  Qb[dst]     = f2bf(a * c - b * s);
  Qb[dst + 1] = f2bf(a * s + b * c);

  size_t ksrc = (size_t)t * (2 * DD) + hh * HDIM + 2 * i;
  a = kvf[ksrc]; b = kvf[ksrc + 1];
  Kb[dst]     = f2bf(a * c - b * s);
  Kb[dst + 1] = f2bf(a * s + b * c);

  size_t vsrc = ksrc + DD;
  Vb[dst]     = f2bf(kvf[vsrc]);
  Vb[dst + 1] = f2bf(kvf[vsrc + 1]);
}

// ---------------------------------------------------------------------------
// Causal flash attention. Grid (TT/64, NHEAD), 128 threads (4 waves).
// Each wave owns 16 query rows; K and V^T tiles staged in LDS per 32-key step.
// Output y written as bf16 [TT, DD] with head-major columns.
// ---------------------------------------------------------------------------
__global__ __launch_bounds__(128) void flash_attn_kernel(
    const unsigned short* __restrict__ Q, const unsigned short* __restrict__ Kmat,
    const unsigned short* __restrict__ V, unsigned short* __restrict__ Y) {
  const int tid  = threadIdx.x;
  const int lane = tid & 31;
  const int wave = tid >> 5;       // 0..3
  const int h    = blockIdx.y;
  const int qb   = blockIdx.x * 64;
  const int qw   = qb + wave * 16; // this wave's query base row
  const int hi   = lane >> 4;
  const int l15  = lane & 15;
  const int koffA = 16 * hi;
  const int koffB = 16 * hi;

  __shared__ __align__(16) unsigned short Kt[32 * 72];      // [k][d]
  __shared__ __align__(16) unsigned short Vt[64 * 40];      // V^T: [d][k]
  __shared__ __align__(16) unsigned short Pt[4][16 * 40];   // per-wave P tile

  const unsigned short* Qh = Q    + (size_t)h * TT * HDIM;
  const unsigned short* Kh = Kmat + (size_t)h * TT * HDIM;
  const unsigned short* Vh = V    + (size_t)h * TT * HDIM;

  // Q fragments: 16 rows x 64 d as two 16x32 A-frags
  Frag qf[2];
#pragma unroll
  for (int f = 0; f < 2; ++f) {
    const unsigned short* qp = Qh + (size_t)(qw + l15) * HDIM + f * 32 + koffA;
    qf[f].q[0] = *(const uint4*)qp;
    qf[f].q[1] = *(const uint4*)(qp + 8);
  }

  float mst[8], lst[8];
  v8f o0 = {}, o1 = {}, o2 = {}, o3 = {};
#pragma unroll
  for (int r = 0; r < 8; ++r) { mst[r] = -INFINITY; lst[r] = 0.0f; }

  const int kend = qb + 64;
  for (int kb = 0; kb < kend; kb += 32) {
    // cooperative stage: K tile [32][64], V^T tile [64][32]
    {
      int rr = tid >> 2;
      int cc = (tid & 3) * 16;
      const uint4* ks = (const uint4*)(Kh + (size_t)(kb + rr) * HDIM + cc);
      *(uint4*)&Kt[rr * 72 + cc]     = ks[0];
      *(uint4*)&Kt[rr * 72 + cc + 8] = ks[1];
      uint4 v0 = *(const uint4*)(Vh + (size_t)(kb + rr) * HDIM + cc);
      uint4 v1 = *(const uint4*)(Vh + (size_t)(kb + rr) * HDIM + cc + 8);
      const unsigned short* vh0 = (const unsigned short*)&v0;
      const unsigned short* vh1 = (const unsigned short*)&v1;
#pragma unroll
      for (int j = 0; j < 8; ++j) {
        Vt[(cc + j) * 40 + rr]     = vh0[j];
        Vt[(cc + 8 + j) * 40 + rr] = vh1[j];
      }
    }
    __syncthreads();

    if (kb <= qw + 15) {  // wave-uniform causal skip
      v8f s0 = {}, s1 = {};
#pragma unroll
      for (int f = 0; f < 2; ++f) {  // K-dim (head dim) chunks of 32
        Frag kf0, kf1;
        const unsigned short* kp0 = &Kt[(0  + l15) * 72 + f * 32 + koffB];
        const unsigned short* kp1 = &Kt[(16 + l15) * 72 + f * 32 + koffB];
        kf0.q[0] = *(const uint4*)kp0; kf0.q[1] = *(const uint4*)(kp0 + 8);
        kf1.q[0] = *(const uint4*)kp1; kf1.q[1] = *(const uint4*)(kp1 + 8);
        s0 = wmma_bf16(qf[f], kf0, s0);
        s1 = wmma_bf16(qf[f], kf1, s1);
      }
      // scale + causal mask + online softmax
      int kj0 = kb + l15;
      int kj1 = kb + 16 + l15;
#pragma unroll
      for (int r = 0; r < 8; ++r) {
        int qi = qw + 8 * hi + r;
        float a0 = s0[r] * 0.125f;
        float a1 = s1[r] * 0.125f;
        a0 = (kj0 > qi) ? -INFINITY : a0;
        a1 = (kj1 > qi) ? -INFINITY : a1;
        float rowm = fmaxf(a0, a1);
#pragma unroll
        for (int m = 8; m >= 1; m >>= 1)
          rowm = fmaxf(rowm, __shfl_xor(rowm, m, 32));
        float mnew = fmaxf(mst[r], rowm);
        float alpha = __expf(mst[r] - mnew);
        mst[r] = mnew;
        float p0 = __expf(a0 - mnew);
        float p1 = __expf(a1 - mnew);
        float rs = p0 + p1;
#pragma unroll
        for (int m = 8; m >= 1; m >>= 1) rs += __shfl_xor(rs, m, 32);
        lst[r] = lst[r] * alpha + rs;
        o0[r] *= alpha; o1[r] *= alpha; o2[r] *= alpha; o3[r] *= alpha;
        int mrow = 8 * hi + r;
        Pt[wave][mrow * 40 + l15]      = f2bf(p0);
        Pt[wave][mrow * 40 + 16 + l15] = f2bf(p1);
      }
      __asm__ __volatile__("" ::: "memory");  // order P stores before reload
      // P(16x32) @ V(32x64)
      Frag pf;
      pf.q[0] = *(const uint4*)&Pt[wave][l15 * 40 + koffA];
      pf.q[1] = *(const uint4*)&Pt[wave][l15 * 40 + koffA + 8];
#pragma unroll
      for (int nb = 0; nb < 4; ++nb) {
        Frag vf;
        const unsigned short* vp = &Vt[(nb * 16 + l15) * 40 + koffB];
        vf.q[0] = *(const uint4*)vp;
        vf.q[1] = *(const uint4*)(vp + 8);
        if (nb == 0) o0 = wmma_bf16(pf, vf, o0);
        else if (nb == 1) o1 = wmma_bf16(pf, vf, o1);
        else if (nb == 2) o2 = wmma_bf16(pf, vf, o2);
        else o3 = wmma_bf16(pf, vf, o3);
      }
    }
    __syncthreads();
  }

#pragma unroll
  for (int r = 0; r < 8; ++r) {
    float inv = 1.0f / lst[r];
    int grow = qw + 8 * hi + r;
    size_t base = (size_t)grow * DD + h * HDIM;
    Y[base + 0 * 16 + l15] = f2bf(o0[r] * inv);
    Y[base + 1 * 16 + l15] = f2bf(o1[r] * inv);
    Y[base + 2 * 16 + l15] = f2bf(o2[r] * inv);
    Y[base + 3 * 16 + l15] = f2bf(o3[r] * inv);
  }
}

// ---------------------------------------------------------------------------
// Router logits: one block per token, one wave per expert.
// ---------------------------------------------------------------------------
__global__ __launch_bounds__(256) void router_kernel(
    const unsigned short* __restrict__ h2, const float* __restrict__ rw,
    float* __restrict__ logits) {
  int t = blockIdx.x;
  int lane = threadIdx.x & 31;
  int e = threadIdx.x >> 5;  // 0..7
  float s = 0.f;
  for (int d = lane; d < DD; d += 32)
    s += bf2f(h2[(size_t)t * DD + d]) * rw[(size_t)d * NE + e];
#pragma unroll
  for (int m = 16; m >= 1; m >>= 1) s += __shfl_xor(s, m, 32);
  if (lane == 0) logits[(size_t)t * NE + e] = s;
}

// ---------------------------------------------------------------------------
// Top-2 + softmax over the 2 selected logits -> dense w_ne [TT][NE]
// ---------------------------------------------------------------------------
__global__ __launch_bounds__(256) void topk_kernel(
    const float* __restrict__ logits, float* __restrict__ wne) {
  int t = blockIdx.x * 256 + threadIdx.x;
  if (t >= TT) return;
  float v[NE];
#pragma unroll
  for (int e = 0; e < NE; ++e) v[e] = logits[(size_t)t * NE + e];
  int i0 = 0;
#pragma unroll
  for (int e = 1; e < NE; ++e) if (v[e] > v[i0]) i0 = e;
  int i1 = (i0 == 0) ? 1 : 0;
#pragma unroll
  for (int e = 0; e < NE; ++e) if (e != i0 && v[e] > v[i1]) i1 = e;
  float p1 = __expf(v[i1] - v[i0]);
  float den = 1.0f + p1;
#pragma unroll
  for (int e = 0; e < NE; ++e) {
    float w = (e == i0) ? (1.0f / den) : ((e == i1) ? (p1 / den) : 0.0f);
    wne[(size_t)t * NE + e] = w;
  }
}

// ---------------------------------------------------------------------------
// SwiGLU activation: act = bf16( silu(g) * u )
// ---------------------------------------------------------------------------
__global__ __launch_bounds__(256) void act_kernel(
    const float* __restrict__ g, const float* __restrict__ u,
    unsigned short* __restrict__ a, size_t n) {
  size_t i = (size_t)blockIdx.x * 256 + threadIdx.x;
  size_t stride = (size_t)gridDim.x * 256;
  for (; i < n; i += stride) {
    float gv = g[i];
    float sg = gv / (1.0f + __expf(-gv));
    a[i] = f2bf(sg * u[i]);
  }
}

// ---------------------------------------------------------------------------
extern "C" void kernel_launch(void* const* d_in, const int* in_sizes, int n_in,
                              void* d_out, int out_size, void* d_ws, size_t ws_size,
                              hipStream_t stream) {
  const float* x    = (const float*)d_in[0];
  const float* fcos = (const float*)d_in[1];
  const float* fsin = (const float*)d_in[2];
  const float* n1w  = (const float*)d_in[3];
  const float* wq   = (const float*)d_in[4];
  const float* bq   = (const float*)d_in[5];
  const float* wkv  = (const float*)d_in[6];
  const float* bkv  = (const float*)d_in[7];
  const float* wo   = (const float*)d_in[8];
  const float* bo   = (const float*)d_in[9];
  const float* n2w  = (const float*)d_in[10];
  const float* rw   = (const float*)d_in[11];
  const float* gw   = (const float*)d_in[12];
  const float* uw   = (const float*)d_in[13];
  const float* dw   = (const float*)d_in[14];
  float* out = (float*)d_out;

  char* ws = (char*)d_ws;
  size_t off = 0;
  auto alloc = [&](size_t bytes) {
    size_t o = off;
    off += (bytes + 255) & ~(size_t)255;
    return o;
  };

  const size_t MB = 1024 * 1024;
  size_t o_wqb  = alloc((size_t)DD * DD * 2);
  size_t o_wkvb = alloc((size_t)DD * 2 * DD * 2);
  size_t o_wob  = alloc((size_t)DD * DD * 2);
  size_t o_gb   = alloc((size_t)NE * DD * HHID * 2);
  size_t o_ub   = alloc((size_t)NE * DD * HHID * 2);
  size_t o_db   = alloc((size_t)NE * HHID * DD * 2);
  size_t o_h1   = alloc((size_t)TT * DD * 2);
  size_t o_h2   = alloc((size_t)TT * DD * 2);
  size_t o_y    = alloc((size_t)TT * DD * 2);
  size_t o_log  = alloc((size_t)TT * NE * 4);
  size_t o_wne  = alloc((size_t)TT * NE * 4);
  size_t o_big  = alloc(40 * MB);  // attention scratch / MoE scratch overlay

  unsigned short* wqb  = (unsigned short*)(ws + o_wqb);
  unsigned short* wkvb = (unsigned short*)(ws + o_wkvb);
  unsigned short* wob  = (unsigned short*)(ws + o_wob);
  unsigned short* gb   = (unsigned short*)(ws + o_gb);
  unsigned short* ub   = (unsigned short*)(ws + o_ub);
  unsigned short* db   = (unsigned short*)(ws + o_db);
  unsigned short* h1   = (unsigned short*)(ws + o_h1);
  unsigned short* h2   = (unsigned short*)(ws + o_h2);
  unsigned short* yb   = (unsigned short*)(ws + o_y);
  float* logits = (float*)(ws + o_log);
  float* wne    = (float*)(ws + o_wne);

  // attention-phase layout inside the big region
  float* qf32           = (float*)(ws + o_big);            //  8 MB [TT,DD]
  float* kvf32          = (float*)(ws + o_big + 8 * MB);   // 16 MB [TT,2DD]
  unsigned short* Qb    = (unsigned short*)(ws + o_big + 24 * MB);  // 4 MB
  unsigned short* Kb    = (unsigned short*)(ws + o_big + 28 * MB);  // 4 MB
  unsigned short* Vb    = (unsigned short*)(ws + o_big + 32 * MB);  // 4 MB
  // MoE-phase layout (attention scratch is dead by then)
  float* gbuf           = (float*)(ws + o_big);            // 16 MB [TT,HHID]
  float* ubuf           = (float*)(ws + o_big + 16 * MB);  // 16 MB
  unsigned short* actb  = (unsigned short*)(ws + o_big + 32 * MB);  // 8 MB

  // 1) weights -> bf16
  f32_to_bf16_kernel<<<2048, 256, 0, stream>>>(wq,  wqb,  (size_t)DD * DD);
  f32_to_bf16_kernel<<<2048, 256, 0, stream>>>(wkv, wkvb, (size_t)DD * 2 * DD);
  f32_to_bf16_kernel<<<2048, 256, 0, stream>>>(wo,  wob,  (size_t)DD * DD);
  f32_to_bf16_kernel<<<4096, 256, 0, stream>>>(gw,  gb,   (size_t)NE * DD * HHID);
  f32_to_bf16_kernel<<<4096, 256, 0, stream>>>(uw,  ub,   (size_t)NE * DD * HHID);
  f32_to_bf16_kernel<<<4096, 256, 0, stream>>>(dw,  db,   (size_t)NE * HHID * DD);

  // 2) h1 = rmsnorm(x)
  rmsnorm_kernel<<<TT, 256, 0, stream>>>(x, n1w, h1);

  // 3) q = h1 @ wq + bq ; kv = h1 @ wkv + bkv
  gemm_bf16<0, false, true, false><<<dim3(DD / 128, TT / 64), 256, 0, stream>>>(
      h1, wqb, bq, nullptr, nullptr, 0, 1.0f, qf32, TT, DD, DD);
  gemm_bf16<0, false, true, false><<<dim3(2 * DD / 128, TT / 64), 256, 0, stream>>>(
      h1, wkvb, bkv, nullptr, nullptr, 0, 1.0f, kvf32, TT, 2 * DD, DD);

  // 4) rope + pack to per-head bf16
  rope_pack_kernel<<<(TT * NHEAD * (HDIM / 2)) / 256, 256, 0, stream>>>(
      qf32, kvf32, fcos, fsin, Qb, Kb, Vb);

  // 5) causal flash attention -> y (bf16)
  flash_attn_kernel<<<dim3(TT / 64, NHEAD), 128, 0, stream>>>(Qb, Kb, Vb, yb);

  // 6) out = x + y @ wo + bo
  gemm_bf16<0, false, true, true><<<dim3(DD / 128, TT / 64), 256, 0, stream>>>(
      yb, wob, bo, x, nullptr, 0, 1.0f, out, TT, DD, DD);

  // 7) h2 = rmsnorm(out)
  rmsnorm_kernel<<<TT, 256, 0, stream>>>(out, n2w, h2);

  // 8) router -> top2 weights
  router_kernel<<<TT, 256, 0, stream>>>(h2, rw, logits);
  topk_kernel<<<TT / 256, 256, 0, stream>>>(logits, wne);

  // 9) MoE experts (tile-skip on routing mask; down-proj accumulates into out)
  const float moe_scale = 0.28867513459481287f;  // 1/sqrt(LAYER_DEPTH=12)
  for (int e = 0; e < NE; ++e) {
    const unsigned short* gwe = gb + (size_t)e * DD * HHID;
    const unsigned short* uwe = ub + (size_t)e * DD * HHID;
    const unsigned short* dwe = db + (size_t)e * HHID * DD;
    gemm_bf16<1, false, false, false><<<dim3(HHID / 128, TT / 64), 256, 0, stream>>>(
        h2, gwe, nullptr, nullptr, wne + e, NE, 1.0f, gbuf, TT, HHID, DD);
    gemm_bf16<1, false, false, false><<<dim3(HHID / 128, TT / 64), 256, 0, stream>>>(
        h2, uwe, nullptr, nullptr, wne + e, NE, 1.0f, ubuf, TT, HHID, DD);
    act_kernel<<<4096, 256, 0, stream>>>(gbuf, ubuf, actb, (size_t)TT * HHID);
    gemm_bf16<2, true, false, false><<<dim3(DD / 128, TT / 64), 256, 0, stream>>>(
        actb, dwe, nullptr, nullptr, wne + e, NE, moe_scale, out, TT, DD, HHID);
  }
}
